// NeuralODEFlow_66589172957246
// MI455X (gfx1250) — matrix operand order repaired and checked by reading, not compiled, not run on hardware
//
#include <hip/hip_runtime.h>
#include <cmath>

typedef _Float16 v8h  __attribute__((ext_vector_type(8)));
typedef _Float16 v16h __attribute__((ext_vector_type(16)));
typedef float    v8f  __attribute__((ext_vector_type(8)));
typedef unsigned int u32;
typedef unsigned long long u64;
typedef u32 u32x4 __attribute__((ext_vector_type(4)));
typedef u32 u32x8 __attribute__((ext_vector_type(8)));

#define DIM     64
#define HID     256
#define MT      64      // batch rows per block
#define NSTEPS  32      // RK4 steps over [0,1]
#define THREADS 256     // 8 wave32

#define HP   264        // hidden-activation pitch (f16) = 256 + 8 (bank skew)
#define INP  72         // MLP-input pitch (f16) = 64 + 8

#define L1K  2          // K-steps (K=32) for layer1 (K=64)
#define L2K  8          // layer2 (K=256)
#define L3K  8          // layer3 (K=256)
#define NT_H 16         // 256/16 column tiles
#define NT_D 4          // 64/16 column tiles
#define FRAG 512        // f16 elements per 32x16 B fragment
#define L1OFF 0
#define L2OFF (L1K*NT_H*FRAG)            // 16384
#define L3OFF (L2OFF + L2K*NT_H*FRAG)    // 81920
#define WTOT  (L3OFF + L3K*NT_D*FRAG)    // 98304 f16 = 192 KB
#define WBYTES (WTOT * 2)                // 196608 bytes

static __device__ __forceinline__ float act_tanh(float x) {
#if __has_builtin(__builtin_amdgcn_tanhf)
  return __builtin_amdgcn_tanhf(x);
#else
  return tanhf(x);
#endif
}

static __device__ __forceinline__ v8f wmma_f16(v16h a, v16h b, v8f c) {
  return __builtin_amdgcn_wmma_f32_16x16x32_f16(false, a, false, b, (short)0, c,
                                                false, false);
}

// A fragment (16x32 f16) from row-major LDS [rows][pitch]
static __device__ __forceinline__ v16h load_a(const _Float16* base, int pitch,
                                              int mt, int ks, int lrow, int lhi) {
  const _Float16* p = base + (mt * 16 + lrow) * pitch + ks * 32 + lhi * 8;
  v8h lo = *(const v8h*)p;
  v8h hi = *(const v8h*)(p + 16);
  return __builtin_shufflevector(lo, hi, 0, 1, 2, 3, 4, 5, 6, 7,
                                 8, 9, 10, 11, 12, 13, 14, 15);
}

// B fragment (32x16 f16), fragment-packed: each lane reads 32 contiguous bytes.
// Works for both LDS (hi weights) and global (streamed lo weights).
template <typename P>
static __device__ __forceinline__ v16h load_b(P fb, int ntiles, int ks, int nt,
                                              int lane) {
  return *(const v16h*)(fb + (ks * ntiles + nt) * FRAG + lane * 16);
}

// value -> (hi, lo) f16 pair in LDS
static __device__ __forceinline__ void store_split(_Float16* hi, _Float16* lo,
                                                   int idx, float v) {
  _Float16 h = (_Float16)v;
  hi[idx] = h;
  lo[idx] = (_Float16)(v - (float)h);
}

// Pre-swizzle f32 weights into split-f16 WMMA B-fragment layout in d_ws.
__global__ __launch_bounds__(THREADS)
void node_pack_kernel(const float* __restrict__ src, int N, int ntiles, int ksteps,
                      _Float16* __restrict__ dhi, _Float16* __restrict__ dlo) {
  int e = blockIdx.x * blockDim.x + threadIdx.x;
  int total = ksteps * ntiles * FRAG;
  if (e >= total) return;
  int frag = e >> 9;
  int slot = e & 511;
  int ks = frag / ntiles, nt = frag % ntiles;
  int idx16 = slot >> 4, kk = slot & 15;
  int n = idx16 & 15, khalf = idx16 >> 4;
  int K  = ks * 32 + khalf * 8 + (kk & 7) + ((kk >> 3) << 4);
  int Nc = nt * 16 + n;
  float w = src[K * N + Nc];
  _Float16 h = (_Float16)w;
  dhi[e] = h;
  dlo[e] = (_Float16)(w - (float)h);
}

__global__ __launch_bounds__(THREADS, 1)
void node_flow_kernel(const float* __restrict__ z0,
                      const float* __restrict__ w1,   // time row w1[64,:]
                      const float* __restrict__ b1,
                      const float* __restrict__ b2,
                      const float* __restrict__ b3,
                      const _Float16* __restrict__ whi,  // packed hi weights (LDS-staged)
                      const _Float16* __restrict__ wlo,  // packed lo weights (L2-streamed)
                      float* __restrict__ out) {
  __shared__ __attribute__((aligned(32))) _Float16 sWhi[WTOT];     // 192 KB hi weights
  __shared__ __attribute__((aligned(32))) _Float16 sHhi[MT * HP];  // h hi
  __shared__ __attribute__((aligned(32))) _Float16 sHlo[MT * HP];  // h lo
  __shared__ __attribute__((aligned(32))) _Float16 sIhi[MT * INP]; // y hi
  __shared__ __attribute__((aligned(32))) _Float16 sIlo[MT * INP]; // y lo
  __shared__ float sB1[HID], sT1[HID], sB2[HID], sB3[DIM];

  const int tid  = threadIdx.x;
  const int lane = tid & 31;
  const int wv   = tid >> 5;
  const int lrow = lane & 15;
  const int lhi  = lane >> 4;
  const int mt   = wv >> 1;   // row-strip 0..3
  const int nh   = wv & 1;    // column half

  // ---- Tensor Data Mover: DMA 192 KB of packed hi weights global -> LDS.
  // D# per CDNA5 ISA: 1-D tile, data_size=8B, 24576 elements.
  if (wv == 0) {
    u32 lds_base = (u32)(size_t)((__attribute__((address_space(3))) const void*)&sWhi[0]);
    u64 ga = (u64)(size_t)whi;
    u32x4 g0;
    g0[0] = 1u;                                        // count=1, user descriptor
    g0[1] = lds_base;                                  // lds_addr
    g0[2] = (u32)ga;                                   // global_addr[31:0]
    g0[3] = (u32)((ga >> 32) & 0x01FFFFFFu) | (2u << 30); // global_addr[56:32], type=2
    const u32 elems = WBYTES / 8;                      // 24576 8-byte elements
    u32x8 g1;
    g1[0] = 3u << 16;                                  // data_size=3 (8B), mask=0
    g1[1] = (elems & 0xFFFFu) << 16;                   // tensor_dim0[15:0] @ bits 63:48
    g1[2] = (elems >> 16) | (1u << 16);                // tensor_dim0[31:16], tensor_dim1=1
    g1[3] = (elems & 0xFFFFu) << 16;                   // tile_dim0 @ bits 127:112
    g1[4] = 0;                                         // tile_dim1/2 unused
    g1[5] = elems;                                     // tensor_dim0_stride[31:0]
    g1[6] = 0;
    g1[7] = 0;
    asm volatile("tensor_load_to_lds %0, %1" :: "s"(g0), "s"(g1) : "memory");
    __builtin_amdgcn_s_wait_tensorcnt(0);
  }

  for (int i = tid; i < HID; i += THREADS) {
    sB1[i] = b1[i];
    sB2[i] = b2[i];
    sT1[i] = w1[64 * HID + i];   // time-feature row of w1
  }
  if (tid < DIM) sB3[tid] = b3[tid];

  // ---- RK4 state lives in registers: this wave exclusively owns output tiles
  // (mt, nh*2) and (mt, nh*2+1); mapping fixed for all steps.
  float zr[2][8];   // z
  float rk[2][8];   // k1 + 2k2 + 2k3 (+ k4)
  {
    const size_t rowBase = (size_t)blockIdx.x * MT;
    #pragma unroll
    for (int ti = 0; ti < 2; ++ti) {
      int nc = (nh * 2 + ti) * 16 + lrow;
      #pragma unroll
      for (int j = 0; j < 8; ++j) {
        int row = mt * 16 + lhi * 8 + j;
        float z = z0[(rowBase + row) * DIM + nc];
        zr[ti][j] = z;
        store_split(sIhi, sIlo, row * INP + nc, z);
      }
    }
  }
  __syncthreads();

  const float hs = 1.0f / (float)NSTEPS;
  #pragma unroll 1
  for (int step = 0; step < NSTEPS; ++step) {
    const float t0 = hs * (float)step;
    #pragma unroll 1
    for (int s = 0; s < 4; ++s) {
      const float ts = t0 + ((s == 0) ? 0.0f : (s == 3) ? hs : 0.5f * hs);

      // ---------- layer 1: y[64x64] @ W1 (+ b1 + t*w1_t) -> tanh -> h[64x256]
      {
        v8f acc[8] = {};
        #pragma unroll
        for (int ks = 0; ks < L1K; ++ks) {
          v16h ah = load_a(sIhi, INP, mt, ks, lrow, lhi);
          v16h al = load_a(sIlo, INP, mt, ks, lrow, lhi);
          #pragma unroll
          for (int i = 0; i < 8; ++i) {
            v16h bh = load_b(sWhi + L1OFF, NT_H, ks, nh * 8 + i, lane);
            v16h bl = load_b(wlo + L1OFF, NT_H, ks, nh * 8 + i, lane);
            acc[i] = wmma_f16(ah, bh, acc[i]);
            acc[i] = wmma_f16(ah, bl, acc[i]);
            acc[i] = wmma_f16(al, bh, acc[i]);
          }
        }
        #pragma unroll
        for (int i = 0; i < 8; ++i) {
          int nc = (nh * 8 + i) * 16 + lrow;
          float bt = sB1[nc] + ts * sT1[nc];
          #pragma unroll
          for (int j = 0; j < 8; ++j)
            store_split(sHhi, sHlo, (mt * 16 + lhi * 8 + j) * HP + nc,
                        act_tanh(acc[i][j] + bt));
        }
      }
      __syncthreads();

      // ---------- layer 2: h @ W2 -> tanh -> h (barrier between read & overwrite)
      {
        v8f acc[8] = {};
        #pragma unroll 2
        for (int ks = 0; ks < L2K; ++ks) {
          v16h ah = load_a(sHhi, HP, mt, ks, lrow, lhi);
          v16h al = load_a(sHlo, HP, mt, ks, lrow, lhi);
          #pragma unroll
          for (int i = 0; i < 8; ++i) {
            v16h bh = load_b(sWhi + L2OFF, NT_H, ks, nh * 8 + i, lane);
            v16h bl = load_b(wlo + L2OFF, NT_H, ks, nh * 8 + i, lane);
            acc[i] = wmma_f16(ah, bh, acc[i]);
            acc[i] = wmma_f16(ah, bl, acc[i]);
            acc[i] = wmma_f16(al, bh, acc[i]);
          }
        }
        __syncthreads();  // all h1 reads complete before overwrite
        #pragma unroll
        for (int i = 0; i < 8; ++i) {
          int nc = (nh * 8 + i) * 16 + lrow;
          float bt = sB2[nc];
          #pragma unroll
          for (int j = 0; j < 8; ++j)
            store_split(sHhi, sHlo, (mt * 16 + lhi * 8 + j) * HP + nc,
                        act_tanh(acc[i][j] + bt));
        }
      }
      __syncthreads();

      // ---------- layer 3: k = h @ W3 + b3, then RK4 stage update (registers)
      {
        v8f c0 = {}, c1 = {};
        #pragma unroll 2
        for (int ks = 0; ks < L3K; ++ks) {
          v16h ah  = load_a(sHhi, HP, mt, ks, lrow, lhi);
          v16h al  = load_a(sHlo, HP, mt, ks, lrow, lhi);
          v16h bh0 = load_b(sWhi + L3OFF, NT_D, ks, nh * 2 + 0, lane);
          v16h bh1 = load_b(sWhi + L3OFF, NT_D, ks, nh * 2 + 1, lane);
          v16h bl0 = load_b(wlo + L3OFF, NT_D, ks, nh * 2 + 0, lane);
          v16h bl1 = load_b(wlo + L3OFF, NT_D, ks, nh * 2 + 1, lane);
          c0 = wmma_f16(ah, bh0, c0);
          c0 = wmma_f16(ah, bl0, c0);
          c0 = wmma_f16(al, bh0, c0);
          c1 = wmma_f16(ah, bh1, c1);
          c1 = wmma_f16(ah, bl1, c1);
          c1 = wmma_f16(al, bh1, c1);
        }
        #pragma unroll
        for (int ti = 0; ti < 2; ++ti) {
          int nc = (nh * 2 + ti) * 16 + lrow;
          float bv = sB3[nc];
          #pragma unroll
          for (int j = 0; j < 8; ++j) {
            float k   = (ti ? c1[j] : c0[j]) + bv;
            int row   = mt * 16 + lhi * 8 + j;
            int iy    = row * INP + nc;
            float y;
            if (s == 0) {
              rk[ti][j] = k;
              y = zr[ti][j] + 0.5f * hs * k;
            } else if (s == 1) {
              rk[ti][j] += 2.0f * k;
              y = zr[ti][j] + 0.5f * hs * k;
            } else if (s == 2) {
              rk[ti][j] += 2.0f * k;
              y = zr[ti][j] + hs * k;
            } else {
              zr[ti][j] += (hs * (1.0f / 6.0f)) * (rk[ti][j] + k);
              y = zr[ti][j];
            }
            store_split(sIhi, sIlo, iy, y);
          }
        }
      }
      __syncthreads();
    }
  }

  { // write z(T): wave-owned elements straight from registers
    const size_t rowBase = (size_t)blockIdx.x * MT;
    #pragma unroll
    for (int ti = 0; ti < 2; ++ti) {
      int nc = (nh * 2 + ti) * 16 + lrow;
      #pragma unroll
      for (int j = 0; j < 8; ++j) {
        int row = mt * 16 + lhi * 8 + j;
        out[(rowBase + row) * DIM + nc] = zr[ti][j];
      }
    }
  }
}

extern "C" void kernel_launch(void* const* d_in, const int* in_sizes, int n_in,
                              void* d_out, int out_size, void* d_ws, size_t ws_size,
                              hipStream_t stream) {
  const float* z0 = (const float*)d_in[0];
  const float* w1 = (const float*)d_in[1];
  const float* b1 = (const float*)d_in[2];
  const float* w2 = (const float*)d_in[3];
  const float* b2 = (const float*)d_in[4];
  const float* w3 = (const float*)d_in[5];
  const float* b3 = (const float*)d_in[6];
  _Float16* whi = (_Float16*)d_ws;          // 192 KB packed hi weights
  _Float16* wlo = whi + WTOT;               // 192 KB packed lo weights

  node_pack_kernel<<<(L1K * NT_H * FRAG + THREADS - 1) / THREADS, THREADS, 0, stream>>>(
      w1, HID, NT_H, L1K, whi + L1OFF, wlo + L1OFF);
  node_pack_kernel<<<(L2K * NT_H * FRAG + THREADS - 1) / THREADS, THREADS, 0, stream>>>(
      w2, HID, NT_H, L2K, whi + L2OFF, wlo + L2OFF);
  node_pack_kernel<<<(L3K * NT_D * FRAG + THREADS - 1) / THREADS, THREADS, 0, stream>>>(
      w3, DIM, NT_D, L3K, whi + L3OFF, wlo + L3OFF);

  int batch = in_sizes[0] / DIM;  // 32768
  node_flow_kernel<<<batch / MT, THREADS, 0, stream>>>(z0, w1, b1, b2, b3, whi, wlo,
                                                       (float*)d_out);
}
